// conv_block_lk_sobel_MoE_22462678958532
// MI455X (gfx1250) — matrix operand order repaired and checked by reading, not compiled, not run on hardware
//
#include <hip/hip_runtime.h>
#include <hip/hip_bf16.h>

typedef __attribute__((ext_vector_type(16))) _Float16 v16h;
typedef __attribute__((ext_vector_type(8)))  _Float16 v8h;
typedef __attribute__((ext_vector_type(8)))  float    v8f;

#define IMG 256
#define INC 32
#define OUTC 64
#define TW 128            // output columns per block
#define XCOLS 136         // x tile cols (ox0-3 .. ox0+132)
#define XROWS 7
#define TMPCOLS 132
#define TMPROWS 5

// ---- workspace byte offsets ----
#define WS_EXPERT 0                         // 8 ints
#define WS_GAP    256                       // 256 floats
#define WS_SCALES 2048                      // 5 convs * (64 scale + 64 shift) floats
#define FRAG_M1   16384                     // 9*4*32*16 f16  = 36864 B
#define FRAG_M2   (FRAG_M1 + 36864)         // 50*4*32*16 f16 = 204800 B
#define FRAG_SX   (FRAG_M2 + 204800)        // 25*4*32*16 f16 = 102400 B
#define FRAG_SY   (FRAG_SX + 102400)
#define FRAG_L    (FRAG_SY + 102400)

#define SMEM_XS_B   (XROWS * XCOLS * INC * 2)            // 60928
#define SMEM_TMP_B  (TMPROWS * TMPCOLS * OUTC * 2)       // 84480
#define SMEM_OUT_B  (TW * OUTC * 4)                      // 32768
#define SMEM_BYTES  (SMEM_XS_B + SMEM_TMP_B + SMEM_OUT_B)

__constant__ float K_SOBX[9] = {1.f,0.f,-1.f, 2.f,0.f,-2.f, 1.f,0.f,-1.f};
__constant__ float K_SOBY[9] = {1.f,2.f,1.f, 0.f,0.f,0.f, -1.f,-2.f,-1.f};
__constant__ float K_LAP[9]  = {0.f,1.f,0.f, 1.f,-4.f,1.f, 0.f,1.f,0.f};

__device__ __forceinline__ v8f wmma_f16(v16h a, v16h b, v8f c) {
  return __builtin_amdgcn_wmma_f32_16x16x32_f16(false, a, false, b, (short)0, c, false, false);
}

__device__ __forceinline__ v8f vzero8() {
  v8f z;
#pragma unroll
  for (int i = 0; i < 8; ++i) z[i] = 0.f;
  return z;
}

// A fragment: 16x32 f16. Lane<16: halves0-15=K0..15.
// Lane>=16: halves0-15=K16..31. pixbase = &src[pixel*ICstride + c0]
__device__ __forceinline__ v16h load_a_frag(const _Float16* pixbase, int lane) {
  int off = (lane & 16) ? 16 : 0;
  v8h lo = *(const v8h*)(pixbase + off);
  v8h hi = *(const v8h*)(pixbase + off + 8);
  v16h a;
#pragma unroll
  for (int i = 0; i < 8; ++i) { a[i] = lo[i]; a[8 + i] = hi[i]; }
  return a;
}

// B fragment: 32x16 f16, pre-swizzled layout [lane][16], lane<16 K0..15, lane>=16 K16..31
__device__ __forceinline__ v16h load_b_frag(const _Float16* p, int lane) {
  const v8h* q = (const v8h*)(p + lane * 16);
  v8h lo = q[0], hi = q[1];
  v16h r;
#pragma unroll
  for (int i = 0; i < 8; ++i) { r[i] = lo[i]; r[8 + i] = hi[i]; }
  return r;
}

// Epilogue helper: BN+ReLU fold and write one accumulator fragment to LDS staging.
template <bool ACCUM>
__device__ __forceinline__ void store_acc(v8f acc, int nt, int m0, int lane,
                                          const float* scale, const float* shift,
                                          float* outs, float wsum) {
  int N = nt * 16 + (lane & 15);
  float s = scale[N], t = shift[N];
  int mh = (lane & 16) ? 8 : 0;
#pragma unroll
  for (int r = 0; r < 8; ++r) {
    int px = m0 + mh + r;
    float v = fmaxf(acc[r] * s + t, 0.f) * wsum;
    int idx = N * TW + px;
    if (ACCUM) outs[idx] += v; else outs[idx] = v;
  }
}

// Depthwise 3x3 into tmp (f16). Writes 0 for out-of-image positions (zero pad for next conv).
__device__ void dw_fill(const _Float16* __restrict__ xs, _Float16* __restrict__ dst,
                        const float* k9, int oy, int ox0, int tid) {
  for (int idx = tid; idx < TMPROWS * TMPCOLS * INC; idx += 256) {
    int ch = idx & 31;
    int c  = (idx >> 5) % TMPCOLS;
    int r  = idx / (32 * TMPCOLS);
    int gy = oy - 2 + r, gx = ox0 - 2 + c;
    float v = 0.f;
    if ((unsigned)gy < (unsigned)IMG && (unsigned)gx < (unsigned)IMG) {
#pragma unroll
      for (int dy = 0; dy < 3; ++dy)
#pragma unroll
        for (int dx = 0; dx < 3; ++dx)
          v += k9[dy * 3 + dx] * (float)xs[((r + dy) * XCOLS + (c + dx)) * INC + ch];
    }
    dst[(r * TMPCOLS + c) * INC + ch] = (_Float16)v;
  }
}

// Generic fused 5x5 conv (IC = 32 or 64) via WMMA, BN+ReLU folded, result into LDS outs.
template <int ICS, bool ACCUM>
__device__ void conv5_stage(const _Float16* __restrict__ src, const _Float16* __restrict__ wf,
                            const float* __restrict__ scale, const float* __restrict__ shift,
                            float* __restrict__ outs, float wsum, int wv, int lane) {
  const int m0 = wv * 16;
  v8f acc0 = vzero8(), acc1 = vzero8(), acc2 = vzero8(), acc3 = vzero8();

  const int nkh = ICS >> 5;   // 1 or 2 K-chunks of 32 per tap
  for (int tap = 0; tap < 25; ++tap) {
    int ky = tap / 5, kx = tap % 5;
    int col = m0 + (lane & 15) + kx;                      // <= 131
    const _Float16* pixbase = src + (ky * TMPCOLS + col) * ICS;
#pragma unroll
    for (int kh = 0; kh < nkh; ++kh) {
      v16h a = load_a_frag(pixbase + kh * 32, lane);
      const _Float16* wkc = wf + (size_t)((tap * nkh + kh) * 4) * 512;
      v16h b0 = load_b_frag(wkc + 0 * 512, lane);
      v16h b1 = load_b_frag(wkc + 1 * 512, lane);
      v16h b2 = load_b_frag(wkc + 2 * 512, lane);
      v16h b3 = load_b_frag(wkc + 3 * 512, lane);
      acc0 = wmma_f16(a, b0, acc0);
      acc1 = wmma_f16(a, b1, acc1);
      acc2 = wmma_f16(a, b2, acc2);
      acc3 = wmma_f16(a, b3, acc3);
    }
  }
  store_acc<ACCUM>(acc0, 0, m0, lane, scale, shift, outs, wsum);
  store_acc<ACCUM>(acc1, 1, m0, lane, scale, shift, outs, wsum);
  store_acc<ACCUM>(acc2, 2, m0, lane, scale, shift, outs, wsum);
  store_acc<ACCUM>(acc3, 3, m0, lane, scale, shift, outs, wsum);
}

// ---------------- gate: GAP ----------------
__global__ void gap_kernel(const float* __restrict__ x, float* __restrict__ gap) {
  __shared__ float red[256];
  int plane = blockIdx.x;                       // b*32 + c
  const float* p = x + (size_t)plane * (IMG * IMG);
  float s = 0.f;
  for (int i = threadIdx.x; i < IMG * IMG; i += 256) s += p[i];
  red[threadIdx.x] = s;
  __syncthreads();
  for (int o = 128; o > 0; o >>= 1) {
    if ((int)threadIdx.x < o) red[threadIdx.x] += red[threadIdx.x + o];
    __syncthreads();
  }
  if (threadIdx.x == 0) gap[plane] = red[0] * (1.0f / (IMG * IMG));
}

// ---------------- gate: MLP + argmax ----------------
__global__ void gate_kernel(const float* gap, const float* w1, const float* b1,
                            const float* bn, const float* w2, const float* b2,
                            int* expert) {
  int b = threadIdx.x;
  if (b >= 8) return;
  float h[32];
  for (int i = 0; i < 32; ++i) {
    float z = b1[i];
    for (int j = 0; j < 32; ++j) z += gap[b * 32 + j] * w1[i * 32 + j];
    float g = bn[i], bb = bn[32 + i], m = bn[64 + i], v = bn[96 + i];
    h[i] = tanhf((z - m) * (g * rsqrtf(v + 1e-5f)) + bb);
  }
  float best = -1e30f; int e = 0;
  for (int o = 0; o < 3; ++o) {
    float k = b2[o];
    for (int j = 0; j < 32; ++j) k += h[j] * w2[o * 32 + j];
    if (k > best) { best = k; e = o; }
  }
  expert[b] = e;
}

// ---------------- prep: BN scale/shift fold (conv bias folded in) ----------------
__global__ void prep_scales(const float* bn, const float* bias, float* scale, float* shift) {
  int o = threadIdx.x;   // 64 threads
  float g = bn[o], b = bn[64 + o], m = bn[128 + o], v = bn[192 + o];
  float s = g * rsqrtf(v + 1e-5f);
  scale[o] = s;
  shift[o] = (bias[o] - m) * s + b;
}

// ---------------- prep: weight -> WMMA B-fragment layout (f16) ----------------
// dst layout: [kc][nt][lane][16]; lane<16: K=j, lane>=16: K=16+j (within 32-chunk)
__global__ void prep_frag(const float* __restrict__ w, _Float16* __restrict__ dst,
                          int IC, int K, int total) {
  int d = blockIdx.x * 256 + threadIdx.x;
  if (d >= total) return;
  int j    = d & 15;
  int lane = (d >> 4) & 31;
  int nt   = (d >> 9) & 3;
  int kc   = d >> 11;
  int per  = IC >> 5;                 // k-chunks per tap
  int tap  = kc / per, kh = kc % per;
  int ch   = kh * 32 + ((lane < 16) ? j : 16 + j);
  int N    = nt * 16 + (lane & 15);
  int ky   = tap / K, kx = tap % K;
  dst[d] = (_Float16)w[((size_t)(N * IC + ch) * K + ky) * K + kx];
}

// ---------------- fused MoE conv block ----------------
__global__ void __launch_bounds__(256)
moe_conv_kernel(const float* __restrict__ x, const char* __restrict__ ws,
                float* __restrict__ out) {
  extern __shared__ char smem[];
  _Float16* xs  = (_Float16*)smem;
  _Float16* tmp = (_Float16*)(smem + SMEM_XS_B);
  float*    outs = (float*)(smem + SMEM_XS_B + SMEM_TMP_B);

  const int tid  = threadIdx.x;
  const int wv   = tid >> 5;
  const int lane = tid & 31;
  const int oxc  = blockIdx.x;       // 0..1
  const int oy   = blockIdx.y;       // 0..255
  const int b    = blockIdx.z;       // 0..7
  const int ox0  = oxc * TW;

  const int e = ((const int*)(ws + WS_EXPERT))[b];
  const float* sc = (const float*)(ws + WS_SCALES);

  // load x tile (7 rows x 136 cols x 32 ch) into LDS as f16, zero outside image
  for (int idx = tid; idx < XROWS * XCOLS * INC; idx += 256) {
    int ch  = idx / (XROWS * XCOLS);
    int rem = idx % (XROWS * XCOLS);
    int row = rem / XCOLS, col = rem % XCOLS;
    int gy = oy - 3 + row, gx = ox0 - 3 + col;
    float v = 0.f;
    if ((unsigned)gy < (unsigned)IMG && (unsigned)gx < (unsigned)IMG)
      v = x[(((size_t)b * INC + ch) * IMG + gy) * IMG + gx];
    xs[(row * XCOLS + col) * INC + ch] = (_Float16)v;
  }
  __syncthreads();

  if (e == 0) {
    // ---- main: conv3x3(32->64)+BN+ReLU into tmp (f16), then conv5x5(64->64) ----
    for (int i = tid; i < (TMPROWS * TMPCOLS * OUTC) / 2; i += 256) ((int*)tmp)[i] = 0;
    __syncthreads();
    const _Float16* wf1 = (const _Float16*)(ws + FRAG_M1);
    const float* s1 = sc + 0 * 128; const float* t1 = s1 + 64;
    for (int mt = wv; mt < 45; mt += 8) {       // 5 rows x 9 col-chunks of 16
      int rA = mt / 9, cc = mt % 9;
      v8f acc0 = vzero8(), acc1 = vzero8(), acc2 = vzero8(), acc3 = vzero8();
      for (int tap = 0; tap < 9; ++tap) {
        int ky = tap / 3, kx = tap % 3;
        int pc = cc * 16 + (lane & 15) + kx;
        if (pc > XCOLS - 1) pc = XCOLS - 1;     // clamp padded-M lanes
        v16h a = load_a_frag(xs + ((rA + ky) * XCOLS + pc) * INC, lane);
        const _Float16* wkc = wf1 + (size_t)(tap * 4) * 512;
        v16h b0 = load_b_frag(wkc + 0 * 512, lane);
        v16h b1 = load_b_frag(wkc + 1 * 512, lane);
        v16h b2 = load_b_frag(wkc + 2 * 512, lane);
        v16h b3 = load_b_frag(wkc + 3 * 512, lane);
        acc0 = wmma_f16(a, b0, acc0);
        acc1 = wmma_f16(a, b1, acc1);
        acc2 = wmma_f16(a, b2, acc2);
        acc3 = wmma_f16(a, b3, acc3);
      }
      int gy = oy - 2 + rA;
      int mh = (lane & 16) ? 8 : 0;
#pragma unroll
      for (int nt = 0; nt < 4; ++nt) {
        v8f acc = (nt == 0) ? acc0 : (nt == 1) ? acc1 : (nt == 2) ? acc2 : acc3;
        int N = nt * 16 + (lane & 15);
        float s = s1[N], t = t1[N];
#pragma unroll
        for (int r = 0; r < 8; ++r) {
          int c = cc * 16 + mh + r;
          int gx = ox0 - 2 + c;
          if (c < TMPCOLS && (unsigned)gy < (unsigned)IMG && (unsigned)gx < (unsigned)IMG) {
            float v = fmaxf(acc[r] * s + t, 0.f);
            tmp[(rA * TMPCOLS + c) * OUTC + N] = (_Float16)v;
          }
        }
      }
    }
    __syncthreads();
    conv5_stage<64, false>(tmp, (const _Float16*)(ws + FRAG_M2),
                           sc + 1 * 128, sc + 1 * 128 + 64, outs, 1.0f, wv, lane);
  } else if (e == 1) {
    // ---- sobel: dwX, dwY -> two conv5x5(32->64), averaged ----
    _Float16* tmp1 = tmp;
    _Float16* tmp2 = tmp + TMPROWS * TMPCOLS * INC;
    dw_fill(xs, tmp1, K_SOBX, oy, ox0, tid);
    dw_fill(xs, tmp2, K_SOBY, oy, ox0, tid);
    __syncthreads();
    conv5_stage<32, false>(tmp1, (const _Float16*)(ws + FRAG_SX),
                           sc + 2 * 128, sc + 2 * 128 + 64, outs, 0.5f, wv, lane);
    conv5_stage<32, true>(tmp2, (const _Float16*)(ws + FRAG_SY),
                          sc + 3 * 128, sc + 3 * 128 + 64, outs, 0.5f, wv, lane);
  } else {
    // ---- laplacian: dwLap -> conv5x5(32->64) ----
    dw_fill(xs, tmp, K_LAP, oy, ox0, tid);
    __syncthreads();
    conv5_stage<32, false>(tmp, (const _Float16*)(ws + FRAG_L),
                           sc + 4 * 128, sc + 4 * 128 + 64, outs, 1.0f, wv, lane);
  }

  __syncthreads();
  // coalesced NCHW store: one output row, 64 ch x 128 cols
  for (int i = tid; i < OUTC * TW; i += 256) {
    int ch = i >> 7, px = i & 127;
    out[(((size_t)b * OUTC + ch) * IMG + oy) * IMG + ox0 + px] = outs[i];
  }
}

extern "C" void kernel_launch(void* const* d_in, const int* in_sizes, int n_in,
                              void* d_out, int out_size, void* d_ws, size_t ws_size,
                              hipStream_t stream) {
  (void)in_sizes; (void)n_in; (void)out_size; (void)ws_size;
  const float* x    = (const float*)d_in[0];
  const float* gw1  = (const float*)d_in[1];
  const float* gb1  = (const float*)d_in[2];
  const float* gbn  = (const float*)d_in[3];
  const float* gw2  = (const float*)d_in[4];
  const float* gb2  = (const float*)d_in[5];
  const float* mw1  = (const float*)d_in[6];
  const float* mb1  = (const float*)d_in[7];
  const float* mbn1 = (const float*)d_in[8];
  const float* mw2  = (const float*)d_in[9];
  const float* mb2  = (const float*)d_in[10];
  const float* mbn2 = (const float*)d_in[11];
  const float* sxw  = (const float*)d_in[12];
  const float* sxb  = (const float*)d_in[13];
  const float* sxbn = (const float*)d_in[14];
  const float* syw  = (const float*)d_in[15];
  const float* syb  = (const float*)d_in[16];
  const float* sybn = (const float*)d_in[17];
  const float* lw   = (const float*)d_in[18];
  const float* lb   = (const float*)d_in[19];
  const float* lbn  = (const float*)d_in[20];

  char* ws = (char*)d_ws;
  int*   expert = (int*)(ws + WS_EXPERT);
  float* gapbuf = (float*)(ws + WS_GAP);
  float* scales = (float*)(ws + WS_SCALES);

  // BN scale/shift folding (conv bias folded into shift)
  prep_scales<<<1, 64, 0, stream>>>(mbn1, mb1, scales + 0 * 128, scales + 0 * 128 + 64);
  prep_scales<<<1, 64, 0, stream>>>(mbn2, mb2, scales + 1 * 128, scales + 1 * 128 + 64);
  prep_scales<<<1, 64, 0, stream>>>(sxbn, sxb, scales + 2 * 128, scales + 2 * 128 + 64);
  prep_scales<<<1, 64, 0, stream>>>(sybn, syb, scales + 3 * 128, scales + 3 * 128 + 64);
  prep_scales<<<1, 64, 0, stream>>>(lbn,  lb,  scales + 4 * 128, scales + 4 * 128 + 64);

  // weight -> WMMA B-fragment swizzle (f16)
  int t_m1 = 9 * 1 * 4 * 32 * 16;
  int t_m2 = 25 * 2 * 4 * 32 * 16;
  int t_s5 = 25 * 1 * 4 * 32 * 16;
  prep_frag<<<(t_m1 + 255) / 256, 256, 0, stream>>>(mw1, (_Float16*)(ws + FRAG_M1), 32, 3, t_m1);
  prep_frag<<<(t_m2 + 255) / 256, 256, 0, stream>>>(mw2, (_Float16*)(ws + FRAG_M2), 64, 5, t_m2);
  prep_frag<<<(t_s5 + 255) / 256, 256, 0, stream>>>(sxw, (_Float16*)(ws + FRAG_SX), 32, 5, t_s5);
  prep_frag<<<(t_s5 + 255) / 256, 256, 0, stream>>>(syw, (_Float16*)(ws + FRAG_SY), 32, 5, t_s5);
  prep_frag<<<(t_s5 + 255) / 256, 256, 0, stream>>>(lw,  (_Float16*)(ws + FRAG_L),  32, 5, t_s5);

  // gate: per-batch expert selection on device
  gap_kernel<<<256, 256, 0, stream>>>(x, gapbuf);
  gate_kernel<<<1, 32, 0, stream>>>(gapbuf, gw1, gb1, gbn, gw2, gb2, expert);

  // fused expert compute (178 KB dynamic LDS — MI455X WGP has 320 KB)
  (void)hipFuncSetAttribute((const void*)moe_conv_kernel,
                            hipFuncAttributeMaxDynamicSharedMemorySize, SMEM_BYTES);
  dim3 grid(2, 256, 8);
  moe_conv_kernel<<<grid, 256, SMEM_BYTES, stream>>>(x, ws, (float*)d_out);
}
